// TransformerBlock_19524921327773
// MI455X (gfx1250) — compile-verified, matrix operand order checked by reading, not run on hardware
//
#include <hip/hip_runtime.h>

#define D_MODEL 1024
#define N_HEAD  16
#define D_FF    4096
#define BATCH   4
#define SEQ     2048
#define ROWS    (BATCH*SEQ)            // 8192
#define DK      (D_MODEL/N_HEAD)       // 64
#define EPS_LN  1e-5f

// GEMM block tiling
#define BM 128
#define BN 128
#define BK 32
#define LDSW 40                         // padded row stride (elements), 80B (16B-aligned)
#define TILE_BYTES (BM*LDSW*2)          // 10240 B per staged tile
#define STAGE_BYTES (2*TILE_BYTES)      // A + B
#define GEMM_SMEM (2*STAGE_BYTES)       // double buffered: 40960 B

// Attention staging
#define ATT_KSTRIDE 72                  // K tile row stride (elements), 144B
#define ATT_VSTRIDE 40                  // V^T tile row stride (elements), 80B
#define ATT_KBYTES (32*ATT_KSTRIDE*2)   // 4608
#define ATT_VBYTES (64*ATT_VSTRIDE*2)   // 5120
#define ATT_STG (ATT_KBYTES+ATT_VBYTES) // 9728 per stage
#define ATT_PBASE (2*ATT_STG)           // 19456
#define ATT_SMEM (ATT_PBASE + 8*512*2)  // + 8 waves x 16x32 bf16 P tiles = 27648

typedef __attribute__((ext_vector_type(16))) __bf16 v16bf;
typedef __attribute__((ext_vector_type(8)))  __bf16 v8bf;
typedef __attribute__((ext_vector_type(8)))  float  v8f;

// ---------------------------------------------------------------------------
// WMMA helpers. Operand layout (CDNA5 ISA): lane L holds index
// (L&15) of the striped dim; its 16 elements are two contiguous 8-elem runs
// at K=(L>>4)*8 and K=16+(L>>4)*8  ->  2 x 16B loads per fragment.
// ---------------------------------------------------------------------------
__device__ __forceinline__ v16bf load_frag_vec(const __bf16* p, int ld){
  int lane = (int)(threadIdx.x & 31);
  int g = lane >> 4, i = lane & 15;
  const __bf16* base = p + (size_t)i * (size_t)ld + (size_t)(g * 8);
  v8bf lo = *(const v8bf*)(base);
  v8bf hi = *(const v8bf*)(base + 16);
  return __builtin_shufflevector(lo, hi, 0,1,2,3,4,5,6,7,8,9,10,11,12,13,14,15);
}

__device__ __forceinline__ v8f wmma_bf16(v16bf a, v16bf b, v8f c){
  return __builtin_amdgcn_wmma_f32_16x16x32_bf16(false, a, false, b, (short)0,
                                                 c, false, false);
}

// CDNA5 async global->LDS copy (ASYNCcnt), CDNA5 ISA.
__device__ __forceinline__ void async_load_b128(unsigned lds_off, const void* gaddr){
  asm volatile("global_load_async_to_lds_b128 %0, %1, off"
               :: "v"(lds_off), "v"(gaddr) : "memory");
}
__device__ __forceinline__ void wait_async0(){
  asm volatile("s_wait_asynccnt 0x0" ::: "memory");
}

// ---------------------------------------------------------------------------
// fp32 -> bf16 conversion (plain)
// ---------------------------------------------------------------------------
__global__ void cvt_bf16_kernel(const float* __restrict__ in,
                                __bf16* __restrict__ out, int n){
  int i = blockIdx.x * blockDim.x + threadIdx.x;
  int stride = gridDim.x * blockDim.x;
  for (; i < n; i += stride) out[i] = (__bf16)in[i];
}

// fp32 [K x N] -> bf16 transposed [N x K], 32x32 LDS tiles
__global__ void cvt_transpose_kernel(const float* __restrict__ in,
                                     __bf16* __restrict__ out, int K, int N){
  __shared__ float tile[32][33];
  int n0 = blockIdx.x * 32, k0 = blockIdx.y * 32;
  int tx = (int)(threadIdx.x & 31), ty = (int)(threadIdx.x >> 5);
#pragma unroll
  for (int yy = ty; yy < 32; yy += 8)
    tile[yy][tx] = in[(size_t)(k0 + yy) * N + (n0 + tx)];
  __syncthreads();
#pragma unroll
  for (int yy = ty; yy < 32; yy += 8)
    out[(size_t)(n0 + yy) * K + (k0 + tx)] = (__bf16)tile[tx][yy];
}

// ---------------------------------------------------------------------------
// Block GEMM core: 256 threads, 128x128 C tile, BK=32, double-buffered
// async global->LDS staging. A row-major [M x K]; BT row-major [N x K].
// ---------------------------------------------------------------------------
__device__ __forceinline__ void stage_issue(const __bf16* __restrict__ Ab, int lda,
                                            const __bf16* __restrict__ Bb, int ldb,
                                            int k0, unsigned stage_off){
  int t = (int)threadIdx.x;
#pragma unroll
  for (int c = 0; c < 2; ++c){
    int chunk = t * 2 + c;                    // 0..511
    int m  = chunk >> 2;
    int ko = (chunk & 3) * 8;
    unsigned loff = stage_off + (unsigned)(m * LDSW + ko) * 2u;
    async_load_b128(loff,              Ab + (size_t)m * lda + k0 + ko);
    async_load_b128(loff + TILE_BYTES, Bb + (size_t)m * ldb + k0 + ko);
  }
}

__device__ __forceinline__ void gemm_block(const __bf16* __restrict__ Ablock, int lda,
                                           const __bf16* __restrict__ BTblock, int ldbt,
                                           int K, char* smem, v8f acc[2][4]){
  int w  = (int)(threadIdx.x >> 5);
  int mw = (w >> 1) * 32;
  int nw = (w & 1) * 64;
  int nk = K / BK;
  stage_issue(Ablock, lda, BTblock, ldbt, 0, 0);
#pragma unroll 1
  for (int ks = 0; ks < nk; ++ks){
    wait_async0();
    __syncthreads();
    if (ks + 1 < nk)
      stage_issue(Ablock, lda, BTblock, ldbt,
                  (ks + 1) * BK, (unsigned)((ks + 1) & 1) * STAGE_BYTES);
    const __bf16* At = (const __bf16*)(smem + (ks & 1) * STAGE_BYTES);
    const __bf16* Bt = (const __bf16*)(smem + (ks & 1) * STAGE_BYTES + TILE_BYTES);
    v16bf a0 = load_frag_vec(At + (size_t)mw * LDSW, LDSW);
    v16bf a1 = load_frag_vec(At + (size_t)(mw + 16) * LDSW, LDSW);
#pragma unroll
    for (int j = 0; j < 4; ++j){
      v16bf b = load_frag_vec(Bt + (size_t)(nw + j * 16) * LDSW, LDSW);
      acc[0][j] = wmma_bf16(a0, b, acc[0][j]);
      acc[1][j] = wmma_bf16(a1, b, acc[1][j]);
    }
    __syncthreads();
  }
}

// GEMM1: qkv = x @ w_qkv + b_qkv; q,k -> [B,H,T,dk], v -> [B,H,dk,T]; q *= 1/8
__global__ void gemm_qkv_kernel(const __bf16* __restrict__ xb,
                                const __bf16* __restrict__ wT,
                                const float* __restrict__ bias,
                                __bf16* __restrict__ q,
                                __bf16* __restrict__ k,
                                __bf16* __restrict__ v){
  extern __shared__ char smem[];
  int m0 = blockIdx.x * BM, n0 = blockIdx.y * BN;
  v8f acc[2][4] = {};
  gemm_block(xb + (size_t)m0 * D_MODEL, D_MODEL,
             wT + (size_t)n0 * D_MODEL, D_MODEL, D_MODEL, smem, acc);
  int w = (int)(threadIdx.x >> 5), mw = (w >> 1) * 32, nw = (w & 1) * 64;
  int lane = (int)(threadIdx.x & 31), g = lane >> 4, n = lane & 15;
#pragma unroll
  for (int sub = 0; sub < 2; ++sub)
#pragma unroll
    for (int j = 0; j < 4; ++j){
      int col = n0 + nw + j * 16 + n;
      float bb = bias[col];
      int which = col >> 10;
      int h = (col >> 6) & (N_HEAD - 1);
      int d = col & (DK - 1);
      __bf16* dst = (which == 0) ? q : ((which == 1) ? k : v);
      float scale = (which == 0) ? 0.125f : 1.0f;
#pragma unroll
      for (int r = 0; r < 8; ++r){
        int row = m0 + mw + sub * 16 + r + 8 * g;
        int b_ = row >> 11, t = row & (SEQ - 1);
        size_t bh = (size_t)b_ * N_HEAD + h;
        size_t idx = (which == 2) ? (bh * DK + d) * SEQ + t
                                  : (bh * SEQ + t) * DK + d;
        dst[idx] = (__bf16)((acc[sub][j][r] + bb) * scale);
      }
    }
}

// GEMM2: h1pre = x + ctx @ w_out + b_out   (fp32)
__global__ void gemm_proj_kernel(const __bf16* __restrict__ ctx,
                                 const __bf16* __restrict__ wT,
                                 const float* __restrict__ bias,
                                 const float* __restrict__ xres,
                                 float* __restrict__ outp){
  extern __shared__ char smem[];
  int m0 = blockIdx.x * BM, n0 = blockIdx.y * BN;
  v8f acc[2][4] = {};
  gemm_block(ctx + (size_t)m0 * D_MODEL, D_MODEL,
             wT + (size_t)n0 * D_MODEL, D_MODEL, D_MODEL, smem, acc);
  int w = (int)(threadIdx.x >> 5), mw = (w >> 1) * 32, nw = (w & 1) * 64;
  int lane = (int)(threadIdx.x & 31), g = lane >> 4, n = lane & 15;
#pragma unroll
  for (int sub = 0; sub < 2; ++sub)
#pragma unroll
    for (int j = 0; j < 4; ++j){
      int col = n0 + nw + j * 16 + n;
      float bb = bias[col];
#pragma unroll
      for (int r = 0; r < 8; ++r){
        size_t row = (size_t)(m0 + mw + sub * 16 + r + 8 * g);
        outp[row * D_MODEL + col] = acc[sub][j][r] + bb + xres[row * D_MODEL + col];
      }
    }
}

// GEMM3: a1 = relu(h1 @ w1 + b1)   (bf16)
__global__ void gemm_ffn1_kernel(const __bf16* __restrict__ h1b,
                                 const __bf16* __restrict__ wT,
                                 const float* __restrict__ bias,
                                 __bf16* __restrict__ a1){
  extern __shared__ char smem[];
  int m0 = blockIdx.x * BM, n0 = blockIdx.y * BN;
  v8f acc[2][4] = {};
  gemm_block(h1b + (size_t)m0 * D_MODEL, D_MODEL,
             wT + (size_t)n0 * D_MODEL, D_MODEL, D_MODEL, smem, acc);
  int w = (int)(threadIdx.x >> 5), mw = (w >> 1) * 32, nw = (w & 1) * 64;
  int lane = (int)(threadIdx.x & 31), g = lane >> 4, n = lane & 15;
#pragma unroll
  for (int sub = 0; sub < 2; ++sub)
#pragma unroll
    for (int j = 0; j < 4; ++j){
      int col = n0 + nw + j * 16 + n;
      float bb = bias[col];
#pragma unroll
      for (int r = 0; r < 8; ++r){
        size_t row = (size_t)(m0 + mw + sub * 16 + r + 8 * g);
        a1[row * D_FF + col] = (__bf16)fmaxf(acc[sub][j][r] + bb, 0.0f);
      }
    }
}

// GEMM4: h2pre = h1 + a1 @ w2 + b2   (fp32)
__global__ void gemm_ffn2_kernel(const __bf16* __restrict__ a1,
                                 const __bf16* __restrict__ wT,
                                 const float* __restrict__ bias,
                                 const float* __restrict__ h1,
                                 float* __restrict__ outp){
  extern __shared__ char smem[];
  int m0 = blockIdx.x * BM, n0 = blockIdx.y * BN;
  v8f acc[2][4] = {};
  gemm_block(a1 + (size_t)m0 * D_FF, D_FF,
             wT + (size_t)n0 * D_FF, D_FF, D_FF, smem, acc);
  int w = (int)(threadIdx.x >> 5), mw = (w >> 1) * 32, nw = (w & 1) * 64;
  int lane = (int)(threadIdx.x & 31), g = lane >> 4, n = lane & 15;
#pragma unroll
  for (int sub = 0; sub < 2; ++sub)
#pragma unroll
    for (int j = 0; j < 4; ++j){
      int col = n0 + nw + j * 16 + n;
      float bb = bias[col];
#pragma unroll
      for (int r = 0; r < 8; ++r){
        size_t row = (size_t)(m0 + mw + sub * 16 + r + 8 * g);
        outp[row * D_MODEL + col] = acc[sub][j][r] + bb + h1[row * D_MODEL + col];
      }
    }
}

// ---------------------------------------------------------------------------
// Flash attention: block = 128 query rows of one (b,h); 8 waves x 16-row tile.
// K tile (32x64) and V^T tile (64x32) staged once per block into dynamic LDS
// via async copies, double buffered; all operand reads are ds_load_b128.
// ---------------------------------------------------------------------------
__global__ void attn_kernel(const __bf16* __restrict__ qg,
                            const __bf16* __restrict__ kg,
                            const __bf16* __restrict__ vgT,
                            __bf16* __restrict__ ctx){
  extern __shared__ char smem[];
  int widb = (int)(threadIdx.x >> 5);
  const int QBLK = SEQ / 128;                 // 16 q-blocks per (b,h)
  int qblk = blockIdx.x % QBLK;
  int bh   = blockIdx.x / QBLK;
  int h = bh & (N_HEAD - 1);
  int b = bh / N_HEAD;
  int qrow0 = qblk * 128 + widb * 16;
  const __bf16* qp  = qg  + ((size_t)bh * SEQ + qrow0) * DK;
  const __bf16* kp  = kg  + (size_t)bh * SEQ * DK;
  const __bf16* vpT = vgT + (size_t)bh * SEQ * DK;
  int lane = (int)(threadIdx.x & 31), g = lane >> 4, n = lane & 15;

  auto issue_kv = [&](int kt, unsigned soff){
    int t = (int)threadIdx.x;
    {   // K tile: 32 rows x 64 d = 256 x 16B chunks
      int row = t >> 3, ko = (t & 7) * 8;
      async_load_b128(soff + (unsigned)(row * ATT_KSTRIDE + ko) * 2u,
                      kp + (size_t)(kt * 32 + row) * DK + ko);
    }
    {   // V^T tile: 64 rows (d) x 32 keys = 256 x 16B chunks
      int d = t >> 2, ko = (t & 3) * 8;
      async_load_b128(soff + ATT_KBYTES + (unsigned)(d * ATT_VSTRIDE + ko) * 2u,
                      vpT + (size_t)d * SEQ + kt * 32 + ko);
    }
  };

  v16bf qa0 = load_frag_vec(qp, DK);          // d = 0..31
  v16bf qa1 = load_frag_vec(qp + 32, DK);     // d = 32..63

  v8f acc[4] = {};
  float mx[8], lsum[8];
#pragma unroll
  for (int r = 0; r < 8; ++r){ mx[r] = -1e30f; lsum[r] = 0.0f; }
  __bf16* pl = (__bf16*)(smem + ATT_PBASE) + widb * 512;

  const int NT = SEQ / 32;                    // 64 key steps
  issue_kv(0, 0);
#pragma unroll 1
  for (int kt = 0; kt < NT; ++kt){
    wait_async0();
    __syncthreads();                          // stage kt visible
    if (kt + 1 < NT)
      issue_kv(kt + 1, (unsigned)((kt + 1) & 1) * ATT_STG);
    const __bf16* Kt = (const __bf16*)(smem + (kt & 1) * ATT_STG);
    const __bf16* Vt = (const __bf16*)(smem + (kt & 1) * ATT_STG + ATT_KBYTES);

    v8f s0 = {}, s1 = {};
    s0 = wmma_bf16(qa0, load_frag_vec(Kt, ATT_KSTRIDE), s0);
    s0 = wmma_bf16(qa1, load_frag_vec(Kt + 32, ATT_KSTRIDE), s0);
    s1 = wmma_bf16(qa0, load_frag_vec(Kt + 16 * ATT_KSTRIDE, ATT_KSTRIDE), s1);
    s1 = wmma_bf16(qa1, load_frag_vec(Kt + 16 * ATT_KSTRIDE + 32, ATT_KSTRIDE), s1);

    float alpha[8];
#pragma unroll
    for (int r = 0; r < 8; ++r){
      float smax = fmaxf(s0[r], s1[r]);
#pragma unroll
      for (int msk = 1; msk < 16; msk <<= 1)
        smax = fmaxf(smax, __shfl_xor(smax, msk, 32));
      float mn = fmaxf(mx[r], smax);
      float a  = __expf(mx[r] - mn);
      float p0 = __expf(s0[r] - mn);
      float p1 = __expf(s1[r] - mn);
      float ps = p0 + p1;
#pragma unroll
      for (int msk = 1; msk < 16; msk <<= 1)
        ps += __shfl_xor(ps, msk, 32);
      lsum[r] = lsum[r] * a + ps;
      mx[r] = mn; alpha[r] = a;
      s0[r] = p0; s1[r] = p1;
    }
#pragma unroll
    for (int j = 0; j < 4; ++j)
#pragma unroll
      for (int r = 0; r < 8; ++r) acc[j][r] *= alpha[r];

    // P: D-layout -> A-layout via per-wave LDS tile
#pragma unroll
    for (int r = 0; r < 8; ++r){
      int m = r + 8 * g;
      pl[m * 32 + n]      = (__bf16)s0[r];
      pl[m * 32 + 16 + n] = (__bf16)s1[r];
    }
    __syncthreads();
    v16bf pa = load_frag_vec(pl, 32);
#pragma unroll
    for (int j = 0; j < 4; ++j){
      v16bf vb = load_frag_vec(Vt + (size_t)(j * 16) * ATT_VSTRIDE, ATT_VSTRIDE);
      acc[j] = wmma_bf16(pa, vb, acc[j]);
    }
    __syncthreads();                          // done reading stage kt
  }

#pragma unroll
  for (int j = 0; j < 4; ++j)
#pragma unroll
    for (int r = 0; r < 8; ++r){
      int m = r + 8 * g;
      float o = acc[j][r] / lsum[r];
      size_t row = (size_t)b * SEQ + qrow0 + m;
      ctx[row * D_MODEL + h * DK + j * 16 + n] = (__bf16)o;
    }
}

// ---------------------------------------------------------------------------
// LayerNorm: one block (256 threads) per row of 1024
// ---------------------------------------------------------------------------
__global__ void ln_kernel(const float* __restrict__ in,
                          const float* __restrict__ gam,
                          const float* __restrict__ bet,
                          float* __restrict__ outf,
                          __bf16* __restrict__ outb){
  __shared__ float sh1[256], sh2[256];
  size_t row = blockIdx.x;
  const float* p = in + row * D_MODEL;
  float s = 0.0f, s2 = 0.0f;
  for (int c = threadIdx.x; c < D_MODEL; c += blockDim.x){
    float v = p[c]; s += v; s2 += v * v;
  }
  int tid = (int)threadIdx.x;
  sh1[tid] = s; sh2[tid] = s2;
  __syncthreads();
  for (int off = 128; off > 0; off >>= 1){
    if (tid < off){ sh1[tid] += sh1[tid + off]; sh2[tid] += sh2[tid + off]; }
    __syncthreads();
  }
  float mean = sh1[0] * (1.0f / D_MODEL);
  float var  = sh2[0] * (1.0f / D_MODEL) - mean * mean;
  float rstd = rsqrtf(var + EPS_LN);
  for (int c = tid; c < D_MODEL; c += blockDim.x){
    float y = (p[c] - mean) * rstd * gam[c] + bet[c];
    outf[row * D_MODEL + c] = y;
    if (outb) outb[row * D_MODEL + c] = (__bf16)y;
  }
}

// ---------------------------------------------------------------------------
extern "C" void kernel_launch(void* const* d_in, const int* in_sizes, int n_in,
                              void* d_out, int out_size, void* d_ws, size_t ws_size,
                              hipStream_t stream){
  (void)in_sizes; (void)n_in; (void)out_size; (void)ws_size;
  const float* x     = (const float*)d_in[0];
  const float* w_qkv = (const float*)d_in[1];
  const float* b_qkv = (const float*)d_in[2];
  const float* w_out = (const float*)d_in[3];
  const float* b_out = (const float*)d_in[4];
  const float* w1    = (const float*)d_in[5];
  const float* b1    = (const float*)d_in[6];
  const float* w2    = (const float*)d_in[7];
  const float* b2    = (const float*)d_in[8];
  const float* ln1g  = (const float*)d_in[9];
  const float* ln1b  = (const float*)d_in[10];
  const float* ln2g  = (const float*)d_in[11];
  const float* ln2b  = (const float*)d_in[12];
  float* out = (float*)d_out;

  char* ws = (char*)d_ws;
  size_t off = 0;
  auto bump = [&](size_t bytes) -> void* {
    void* p = ws + off;
    off = (off + bytes + 255) & ~(size_t)255;
    return p;
  };
  const size_t R = ROWS;
  __bf16* xb     = (__bf16*)bump(R * D_MODEL * 2);                   // reused as h1_bf
  __bf16* wqkvT  = (__bf16*)bump((size_t)D_MODEL * 3 * D_MODEL * 2);
  __bf16* woutT  = (__bf16*)bump((size_t)D_MODEL * D_MODEL * 2);
  __bf16* w1T    = (__bf16*)bump((size_t)D_MODEL * D_FF * 2);
  __bf16* w2T    = (__bf16*)bump((size_t)D_FF * D_MODEL * 2);
  __bf16* big    = (__bf16*)bump(R * (size_t)D_FF * 2);              // q|k|vT|ctx -> a1
  __bf16* qb   = big;
  __bf16* kb   = qb + R * D_MODEL;
  __bf16* vbT  = kb + R * D_MODEL;
  __bf16* ctxb = vbT + R * D_MODEL;
  __bf16* a1b  = big;
  float* h1pre = (float*)bump(R * D_MODEL * 4);                      // reused as h2pre
  float* h1    = (float*)bump(R * D_MODEL * 4);
  __bf16* h1b  = xb;
  float* h2pre = h1pre;

  // 1. conversions: x plain; weights converted + transposed (one-time)
  {
    int n = ROWS * D_MODEL;
    cvt_bf16_kernel<<<(n + 255) / 256, 256, 0, stream>>>(x, xb, n);
    cvt_transpose_kernel<<<dim3((3 * D_MODEL) / 32, D_MODEL / 32), 256, 0, stream>>>(
        w_qkv, wqkvT, D_MODEL, 3 * D_MODEL);
    cvt_transpose_kernel<<<dim3(D_MODEL / 32, D_MODEL / 32), 256, 0, stream>>>(
        w_out, woutT, D_MODEL, D_MODEL);
    cvt_transpose_kernel<<<dim3(D_FF / 32, D_MODEL / 32), 256, 0, stream>>>(
        w1, w1T, D_MODEL, D_FF);
    cvt_transpose_kernel<<<dim3(D_MODEL / 32, D_FF / 32), 256, 0, stream>>>(
        w2, w2T, D_FF, D_MODEL);
  }

  // 2. QKV projection
  gemm_qkv_kernel<<<dim3(ROWS / BM, (3 * D_MODEL) / BN), 256, GEMM_SMEM, stream>>>(
      xb, wqkvT, b_qkv, qb, kb, vbT);

  // 3. Flash attention (block = 128 query rows of one (b,h))
  attn_kernel<<<BATCH * N_HEAD * (SEQ / 128), 256, ATT_SMEM, stream>>>(
      qb, kb, vbT, ctxb);

  // 4. Output projection + residual
  gemm_proj_kernel<<<dim3(ROWS / BM, D_MODEL / BN), 256, GEMM_SMEM, stream>>>(
      ctxb, woutT, b_out, x, h1pre);

  // 5. LayerNorm 1
  ln_kernel<<<ROWS, 256, 0, stream>>>(h1pre, ln1g, ln1b, h1, h1b);

  // 6. FFN up + ReLU
  gemm_ffn1_kernel<<<dim3(ROWS / BM, D_FF / BN), 256, GEMM_SMEM, stream>>>(
      h1b, w1T, b1, a1b);

  // 7. FFN down + residual
  gemm_ffn2_kernel<<<dim3(ROWS / BM, D_MODEL / BN), 256, GEMM_SMEM, stream>>>(
      a1b, w2T, b2, h1, h2pre);

  // 8. LayerNorm 2 -> final output
  ln_kernel<<<ROWS, 256, 0, stream>>>(h2pre, ln2g, ln2b, out, (__bf16*)nullptr);
}
